// DecoderBlock_11828339933832
// MI455X (gfx1250) — matrix operand order repaired and checked by reading, not compiled, not run on hardware
//
#include <hip/hip_runtime.h>
#include <hip/hip_bf16.h>

// ---------------------------------------------------------------------------
// Decoder block for MI455X (gfx1250, wave32, WMMA).
// GEMMs: v_wmma_f32_16x16x32_bf16, bf16 operands / f32 accumulate.
// Weights transposed+converted to bf16 [N][K] once per call; GEMM k-tiles are
// double-buffered in LDS and filled with global_load_async_to_lds_b128
// (ASYNCcnt) so HBM latency hides behind WMMA compute.
// ---------------------------------------------------------------------------

#define USE_ASYNC_LDS 1

typedef __bf16 bf16_t;
typedef __attribute__((ext_vector_type(8)))  bf16_t bf16x8;
typedef __attribute__((ext_vector_type(16))) bf16_t bf16x16;
typedef __attribute__((ext_vector_type(8)))  float  f32x8;
typedef __attribute__((address_space(3))) bf16_t* lds_p;

static constexpr int Bc = 4, S = 1024, E = 1024, H = 16, KD = 64, D = 4096;
static constexpr int BS = Bc * S;   // 4096 rows (b,s)
static constexpr int HB = H * Bc;   // 64 (head,batch) pairs

// 16 contiguous bf16 -> fragment (B-operand: lane holds K = k0..k0+15)
__device__ __forceinline__ bf16x16 ld_frag_contig(const bf16_t* p) {
  bf16x8 a = *(const bf16x8*)p;
  bf16x8 b = *(const bf16x8*)(p + 8);
  return __builtin_shufflevector(a, b, 0,1,2,3,4,5,6,7,8,9,10,11,12,13,14,15);
}
// A-operand: per-lane two 8-element runs separated by 16 (K interleave)
__device__ __forceinline__ bf16x16 ld_frag_a(const bf16_t* p) {
  bf16x8 a = *(const bf16x8*)p;
  bf16x8 b = *(const bf16x8*)(p + 16);
  return __builtin_shufflevector(a, b, 0,1,2,3,4,5,6,7,8,9,10,11,12,13,14,15);
}
__device__ __forceinline__ f32x8 wmma_bf16(bf16x16 a, bf16x16 b, f32x8 c) {
  return __builtin_amdgcn_wmma_f32_16x16x32_bf16(false, a, false, b,
                                                 (short)0, c, false, false);
}

// ---------------------------------------------------------------------------
// LayerNorm: fp32 [rows, n] -> bf16 [rows, n]
// ---------------------------------------------------------------------------
__global__ __launch_bounds__(256)
void ln_kernel(const float* __restrict__ x, const float* __restrict__ g,
               const float* __restrict__ b, bf16_t* __restrict__ out, int n) {
  const int row = blockIdx.x;
  const float* xr = x + (size_t)row * n;
  float s = 0.f, s2 = 0.f;
  for (int i = threadIdx.x; i < n; i += 256) { float v = xr[i]; s += v; s2 += v * v; }
#pragma unroll
  for (int off = 16; off; off >>= 1) {
    s  += __shfl_xor(s,  off, 32);
    s2 += __shfl_xor(s2, off, 32);
  }
  __shared__ float sh[8], sh2[8];
  if ((threadIdx.x & 31) == 0) { sh[threadIdx.x >> 5] = s; sh2[threadIdx.x >> 5] = s2; }
  __syncthreads();
  s = 0.f; s2 = 0.f;
#pragma unroll
  for (int i = 0; i < 8; i++) { s += sh[i]; s2 += sh2[i]; }
  const float mean = s / n;
  const float rstd = rsqrtf(s2 / n - mean * mean + 1e-5f);
  bf16_t* orow = out + (size_t)row * n;
  for (int i = threadIdx.x; i < n; i += 256)
    orow[i] = (bf16_t)((xr[i] - mean) * rstd * g[i] + b[i]);
}

// ---------------------------------------------------------------------------
// Weight transpose+convert: in f32 [K][N] (ldin) -> out bf16 [N][K] (ldout).
// ---------------------------------------------------------------------------
__global__ __launch_bounds__(256)
void wtrans_kernel(const float* __restrict__ in, int ldin, long sIn,
                   bf16_t* __restrict__ out, int ldout, long sOut) {
  __shared__ bf16_t t[64 * 72];
  const int k0 = blockIdx.x * 64, n0 = blockIdx.y * 64, z = blockIdx.z;
  const float* ib = in + (size_t)z * sIn;
  bf16_t* ob = out + (size_t)z * sOut;
  const int tid = threadIdx.x;
#pragma unroll
  for (int i = 0; i < 4; i++) {
    const int c = tid + 256 * i;
    const int r = c >> 4, cc = (c & 15) * 4;
    float4 v = *(const float4*)&ib[(size_t)(k0 + r) * ldin + n0 + cc];
    bf16_t* tp = &t[r * 72 + cc];
    tp[0] = (bf16_t)v.x; tp[1] = (bf16_t)v.y;
    tp[2] = (bf16_t)v.z; tp[3] = (bf16_t)v.w;
  }
  __syncthreads();
#pragma unroll
  for (int i = 0; i < 2; i++) {
    const int c = tid + 256 * i;
    const int n = c >> 3, j = (c & 7) * 8;
    bf16x8 o;
#pragma unroll
    for (int jj = 0; jj < 8; jj++) o[jj] = t[(j + jj) * 72 + n];
    *(bf16x8*)&ob[(size_t)(n0 + n) * ldout + k0 + j] = o;
  }
}

// ---------------------------------------------------------------------------
// WMMA GEMM: C[M,N] = A[M,K](bf16,row) * Bt[N,K](bf16,row)^T
//            (+bias f32)(+relu)(+residual f32)
// BM=128, BN=128, BK=32; 8 waves as 4x2, each a 32x64 C panel = 8 WMMA/kstep.
// Double-buffered LDS filled by global_load_async_to_lds_b128; next tile
// streams in while the current tile is consumed by WMMAs.
// ---------------------------------------------------------------------------
template <bool RELU, bool OUTBF>
__global__ __launch_bounds__(256)
void gemm_kernel(const bf16_t* __restrict__ A, int lda,
                 const bf16_t* __restrict__ Bt, int ldb,
                 void* __restrict__ C, int ldc,
                 const float* __restrict__ bias,
                 const float* __restrict__ res, int K) {
  __shared__ bf16_t As[2][128 * 40];  // 10240 B per buffer, padded rows
  __shared__ bf16_t Bs[2][128 * 40];
  const int tid = threadIdx.x;
  const int m0 = blockIdx.y * 128, n0 = blockIdx.x * 128;
  const int w = tid >> 5, lane = tid & 31, ql = lane & 15, half = lane >> 4;
  const int wm = (w & 3) * 32, wn = (w >> 2) * 64;
  const int row = tid >> 2, col = (tid & 3) * 8;  // staging: 8 bf16 per chunk
  f32x8 acc[2][4] = {};

  const bf16_t* ap0 = &A [(size_t)(m0 + row)      * lda + col];
  const bf16_t* ap1 = &A [(size_t)(m0 + row + 64) * lda + col];
  const bf16_t* bp0 = &Bt[(size_t)(n0 + row)      * ldb + col];
  const bf16_t* bp1 = &Bt[(size_t)(n0 + row + 64) * ldb + col];

#if USE_ASYNC_LDS
  const unsigned la = (unsigned)(size_t)(lds_p)&As[0][row * 40 + col];
  const unsigned lb = (unsigned)(size_t)(lds_p)&Bs[0][row * 40 + col];
  auto issue = [&](int kk, int buf) {
    const unsigned a = la + (unsigned)buf * 10240u;
    const unsigned b = lb + (unsigned)buf * 10240u;
    asm volatile("global_load_async_to_lds_b128 %0, %1, off"
                 :: "v"(a),         "v"((const void*)(ap0 + kk)) : "memory");
    asm volatile("global_load_async_to_lds_b128 %0, %1, off"
                 :: "v"(a + 5120u), "v"((const void*)(ap1 + kk)) : "memory");
    asm volatile("global_load_async_to_lds_b128 %0, %1, off"
                 :: "v"(b),         "v"((const void*)(bp0 + kk)) : "memory");
    asm volatile("global_load_async_to_lds_b128 %0, %1, off"
                 :: "v"(b + 5120u), "v"((const void*)(bp1 + kk)) : "memory");
  };
  issue(0, 0);
  asm volatile("s_wait_asynccnt 0" ::: "memory");
  __syncthreads();
#else
  bf16x8 ar0 = *(const bf16x8*)ap0, ar1 = *(const bf16x8*)ap1;
  bf16x8 br0 = *(const bf16x8*)bp0, br1 = *(const bf16x8*)bp1;
#endif

  const int nt = K >> 5;
  for (int t = 0; t < nt; t++) {
    const int buf = t & 1;
#if USE_ASYNC_LDS
    if (t + 1 < nt) issue((t + 1) << 5, 1 - buf);
#else
    __syncthreads();
    *(bf16x8*)&As[buf][row * 40 + col]        = ar0;
    *(bf16x8*)&As[buf][(row + 64) * 40 + col] = ar1;
    *(bf16x8*)&Bs[buf][row * 40 + col]        = br0;
    *(bf16x8*)&Bs[buf][(row + 64) * 40 + col] = br1;
    if (t + 1 < nt) {
      const int kk = (t + 1) << 5;
      ar0 = *(const bf16x8*)(ap0 + kk);
      ar1 = *(const bf16x8*)(ap1 + kk);
      br0 = *(const bf16x8*)(bp0 + kk);
      br1 = *(const bf16x8*)(bp1 + kk);
    }
    __syncthreads();
#endif
    if (t + 2 < nt) {                      // L2 prefetch two tiles ahead
      __builtin_prefetch(ap0 + ((t + 2) << 5), 0, 1);
      __builtin_prefetch(bp0 + ((t + 2) << 5), 0, 1);
    }

    bf16x16 af[2], bfr[4];
#pragma unroll
    for (int i = 0; i < 2; i++)
      af[i] = ld_frag_contig(&As[buf][(wm + i * 16 + ql) * 40 + half * 16]);
#pragma unroll
    for (int j = 0; j < 4; j++)
      bfr[j] = ld_frag_contig(&Bs[buf][(wn + j * 16 + ql) * 40 + half * 16]);
#pragma unroll
    for (int i = 0; i < 2; i++)
#pragma unroll
      for (int j = 0; j < 4; j++)
        acc[i][j] = wmma_bf16(af[i], bfr[j], acc[i][j]);

#if USE_ASYNC_LDS
    if (t + 1 < nt) {
      asm volatile("s_wait_asynccnt 0" ::: "memory");
      __syncthreads();   // emits s_wait_dscnt before barrier: WAR-safe
    }
#endif
  }

#pragma unroll
  for (int i = 0; i < 2; i++)
#pragma unroll
    for (int j = 0; j < 4; j++)
#pragma unroll
      for (int r = 0; r < 8; r++) {
        const int gm = m0 + wm + i * 16 + r + 8 * half;
        const int gn = n0 + wn + j * 16 + ql;
        float v = acc[i][j][r];
        if (bias) v += bias[gn];
        if (RELU) v = fmaxf(v, 0.f);
        if (res)  v += res[(size_t)gm * ldc + gn];
        const size_t o = (size_t)gm * ldc + gn;
        if (OUTBF) ((bf16_t*)C)[o] = (bf16_t)v;
        else       ((float*)C)[o]  = v;
      }
}

// ---------------------------------------------------------------------------
// V transpose: vcat [B,S,E] bf16 (head h cols h*64..) -> vt [HB][64][S] bf16
// ---------------------------------------------------------------------------
__global__ __launch_bounds__(256)
void vtrans_kernel(const bf16_t* __restrict__ v, bf16_t* __restrict__ vt) {
  __shared__ bf16_t t[64 * 72];
  const int hb = blockIdx.y, s0 = blockIdx.x * 64, tid = threadIdx.x;
  const int h = hb >> 2, b = hb & 3;
  const bf16_t* vb = v + (size_t)(b * S + s0) * E + h * KD;
  bf16_t* ob = vt + (size_t)hb * KD * S;
#pragma unroll
  for (int i = 0; i < 2; i++) {
    const int c = tid + 256 * i, r = c >> 3, col = (c & 7) * 8;
    *(bf16x8*)&t[r * 72 + col] = *(const bf16x8*)&vb[(size_t)r * E + col];
  }
  __syncthreads();
#pragma unroll
  for (int i = 0; i < 2; i++) {
    const int c = tid + 256 * i, vd = c >> 3, j = (c & 7) * 8;
    bf16x8 o;
#pragma unroll
    for (int jj = 0; jj < 8; jj++) o[jj] = t[(j + jj) * 72 + vd];
    *(bf16x8*)&ob[(size_t)vd * S + s0 + j] = o;
  }
}

// ---------------------------------------------------------------------------
// Flash attention helpers
// ---------------------------------------------------------------------------
__device__ __forceinline__ void load_kv32(const bf16_t* kg, const bf16_t* vg,
                                          int kb, int ql, int half,
                                          bf16x16* kf, bf16x16* vf) {
  const bf16_t* kr0 = kg + (size_t)(kb + ql) * E + half * 16;
  const bf16_t* kr1 = kg + (size_t)(kb + 16 + ql) * E + half * 16;
  kf[0] = ld_frag_contig(kr0);
  kf[1] = ld_frag_contig(kr0 + 32);
  kf[2] = ld_frag_contig(kr1);
  kf[3] = ld_frag_contig(kr1 + 32);
#pragma unroll
  for (int st = 0; st < 4; st++)
    vf[st] = ld_frag_contig(vg + (size_t)(st * 16 + ql) * S + kb + half * 16);
}

template <bool MASK>
__device__ __forceinline__ void attn_step(int kb, int qidx, int half,
                                          const bf16x16* kf, const bf16x16* vf,
                                          const bf16x16& qf0, const bf16x16& qf1,
                                          f32x8* o, float& m_run, float& l_run) {
  f32x8 c0 = {}, c1 = {};
  c0 = wmma_bf16(kf[0], qf0, c0);
  c0 = wmma_bf16(kf[1], qf1, c0);
  c1 = wmma_bf16(kf[2], qf0, c1);
  c1 = wmma_bf16(kf[3], qf1, c1);

  float mx = -1e30f;
#pragma unroll
  for (int r = 0; r < 8; r++) {
    float s0 = c0[r] * 0.125f;
    float s1 = c1[r] * 0.125f;
    if (MASK) {
      const int k0 = kb + r + 8 * half;
      s0 = (k0      <= qidx) ? s0 : -1e30f;
      s1 = (k0 + 16 <= qidx) ? s1 : -1e30f;
    }
    c0[r] = s0; c1[r] = s1;
    mx = fmaxf(mx, fmaxf(s0, s1));
  }
  mx = fmaxf(mx, __shfl_xor(mx, 16, 32));
  const float m_new = fmaxf(m_run, mx);
  const float scale = __expf(m_run - m_new);
  float lsum = 0.f;
#pragma unroll
  for (int r = 0; r < 8; r++) {
    const float e0 = __expf(c0[r] - m_new);
    const float e1 = __expf(c1[r] - m_new);
    c0[r] = e0; c1[r] = e1;
    lsum += e0 + e1;
  }
  lsum += __shfl_xor(lsum, 16, 32);
  l_run = l_run * scale + lsum;
  m_run = m_new;
#pragma unroll
  for (int t = 0; t < 4; t++)
#pragma unroll
    for (int r = 0; r < 8; r++) o[t][r] *= scale;

  // P^T B-fragment: half the keys come from the lane^16 partner.
  float x0[8], x1[8];
#pragma unroll
  for (int r = 0; r < 8; r++) {
    x0[r] = __shfl_xor(c0[r], 16, 32);
    x1[r] = __shfl_xor(c1[r], 16, 32);
  }
  bf16x16 pf;
#pragma unroll
  for (int jj = 0; jj < 8; jj++) {
    pf[jj]     = (bf16_t)(half ? x1[jj] : c0[jj]);
    pf[8 + jj] = (bf16_t)(half ? c1[jj] : x0[jj]);
  }
#pragma unroll
  for (int st = 0; st < 4; st++)
    o[st] = wmma_bf16(vf[st], pf, o[st]);
}

// ---------------------------------------------------------------------------
// Causal flash attention (WMMA): S^T = K*Q^T, O^T = V^T*P^T.
// Main loop: 64-key chunks, no masking, all 16 fragments issued before the
// first WMMA (block-1 loads overlap block-0 compute). Masked tail covers the
// last <=3 key-blocks. One wave = 16 queries; grid (S/128, H*B).
// ---------------------------------------------------------------------------
__global__ __launch_bounds__(256)
void attn_kernel(const bf16_t* __restrict__ q, const bf16_t* __restrict__ k,
                 const bf16_t* __restrict__ vt, bf16_t* __restrict__ ocat) {
  const int hb = blockIdx.y;
  const int w = threadIdx.x >> 5, lane = threadIdx.x & 31;
  const int ql = lane & 15, half = lane >> 4;
  const int qb = blockIdx.x * 128 + w * 16;
  const int h = hb >> 2, b = hb & 3;
  const bf16_t* qg = q  + (size_t)(b * S) * E + h * KD;
  const bf16_t* kg = k  + (size_t)(b * S) * E + h * KD;
  const bf16_t* vg = vt + (size_t)hb * KD * S;

  const bf16_t* qr = qg + (size_t)(qb + ql) * E + half * 16;
  const bf16x16 qf0 = ld_frag_contig(qr);
  const bf16x16 qf1 = ld_frag_contig(qr + 32);

  f32x8 o[4] = {};
  float m_run = -1e30f, l_run = 0.f;
  const int qidx = qb + ql;

  int kmain = qb - 31;                 // keys < kmain are maskless for all lanes
  if (kmain < 0) kmain = 0;
  kmain &= ~63;

  int kb = 0;
  for (; kb < kmain; kb += 64) {       // unmasked main loop, 64 keys/iter
    bf16x16 kfA[4], vfA[4], kfB[4], vfB[4];
    load_kv32(kg, vg, kb,      ql, half, kfA, vfA);
    load_kv32(kg, vg, kb + 32, ql, half, kfB, vfB);
    attn_step<false>(kb,      qidx, half, kfA, vfA, qf0, qf1, o, m_run, l_run);
    attn_step<false>(kb + 32, qidx, half, kfB, vfB, qf0, qf1, o, m_run, l_run);
  }
  for (; kb < qb + 16; kb += 32) {     // masked causal tail
    bf16x16 kf[4], vf[4];
    load_kv32(kg, vg, kb, ql, half, kf, vf);
    attn_step<true>(kb, qidx, half, kf, vf, qf0, qf1, o, m_run, l_run);
  }

  const float inv = 1.f / l_run;
  bf16_t* ob = ocat + (size_t)(b * S + qb + ql) * E + h * KD;
#pragma unroll
  for (int st = 0; st < 4; st++)
#pragma unroll
    for (int r = 0; r < 8; r++) {
      const int vd = st * 16 + r + 8 * half;
      ob[vd] = (bf16_t)(o[st][r] * inv);
    }
}

// ---------------------------------------------------------------------------
extern "C" void kernel_launch(void* const* d_in, const int* in_sizes, int n_in,
                              void* d_out, int out_size, void* d_ws, size_t ws_size,
                              hipStream_t stream) {
  const float* x     = (const float*)d_in[0];
  const float* ln1_g = (const float*)d_in[1];
  const float* ln1_b = (const float*)d_in[2];
  const float* Wq    = (const float*)d_in[3];
  const float* Wk    = (const float*)d_in[4];
  const float* Wv    = (const float*)d_in[5];
  const float* projW = (const float*)d_in[6];
  const float* projb = (const float*)d_in[7];
  const float* ln2_g = (const float*)d_in[8];
  const float* ln2_b = (const float*)d_in[9];
  const float* finW  = (const float*)d_in[10];
  const float* finb  = (const float*)d_in[11];
  const float* hidW  = (const float*)d_in[12];
  const float* hidb  = (const float*)d_in[13];
  const float* foutW = (const float*)d_in[14];
  const float* foutb = (const float*)d_in[15];
  float* out = (float*)d_out;

  char* ws = (char*)d_ws;
  size_t off = 0;
  auto alloc = [&](size_t bytes) -> void* {
    void* p = ws + off;
    off += (bytes + 255) & ~(size_t)255;
    return p;
  };
  // activations
  bf16_t* h1   = (bf16_t*)alloc((size_t)BS * E * 2);
  bf16_t* qcat = (bf16_t*)alloc((size_t)BS * E * 2);
  bf16_t* kcat = (bf16_t*)alloc((size_t)BS * E * 2);
  bf16_t* vcat = (bf16_t*)alloc((size_t)BS * E * 2);
  bf16_t* vtb  = (bf16_t*)alloc((size_t)HB * KD * S * 2);
  bf16_t* ocat = (bf16_t*)alloc((size_t)BS * E * 2);
  float*  x1   = (float*) alloc((size_t)BS * E * 4);
  bf16_t* h2   = (bf16_t*)alloc((size_t)BS * E * 2);
  bf16_t* f0   = (bf16_t*)alloc((size_t)BS * D * 2);
  bf16_t* f1   = (bf16_t*)alloc((size_t)BS * D * 2);
  // transposed bf16 weights [N][K]
  bf16_t* Wqt  = (bf16_t*)alloc((size_t)E * E * 2);
  bf16_t* Wkt  = (bf16_t*)alloc((size_t)E * E * 2);
  bf16_t* Wvt  = (bf16_t*)alloc((size_t)E * E * 2);
  bf16_t* pWt  = (bf16_t*)alloc((size_t)E * E * 2);
  bf16_t* fiWt = (bf16_t*)alloc((size_t)D * E * 2);
  bf16_t* hWt  = (bf16_t*)alloc((size_t)2 * D * D * 2);
  bf16_t* foWt = (bf16_t*)alloc((size_t)E * D * 2);

  // 0) weight transpose/convert (one streaming HBM pass)
  wtrans_kernel<<<dim3(16, 1, 16), 256, 0, stream>>>(Wq, KD, (long)E * KD, Wqt, E, (long)KD * E);
  wtrans_kernel<<<dim3(16, 1, 16), 256, 0, stream>>>(Wk, KD, (long)E * KD, Wkt, E, (long)KD * E);
  wtrans_kernel<<<dim3(16, 1, 16), 256, 0, stream>>>(Wv, KD, (long)E * KD, Wvt, E, (long)KD * E);
  wtrans_kernel<<<dim3(16, 16, 1), 256, 0, stream>>>(projW, E, 0, pWt, E, 0);
  wtrans_kernel<<<dim3(16, 64, 1), 256, 0, stream>>>(finW, D, 0, fiWt, E, 0);
  wtrans_kernel<<<dim3(64, 64, 2), 256, 0, stream>>>(hidW, D, (long)D * D, hWt, D, (long)D * D);
  wtrans_kernel<<<dim3(64, 16, 1), 256, 0, stream>>>(foutW, E, 0, foWt, D, 0);

  // 1) LN1: x -> h1 (bf16)
  ln_kernel<<<BS, 256, 0, stream>>>(x, ln1_g, ln1_b, h1, E);

  // 2) QKV projections into head-concatenated [B,S,E] layout
  const dim3 g1k(E / 128, BS / 128, 1);
  gemm_kernel<false, true><<<g1k, 256, 0, stream>>>(h1, E, Wqt, E, qcat, E, nullptr, nullptr, E);
  gemm_kernel<false, true><<<g1k, 256, 0, stream>>>(h1, E, Wkt, E, kcat, E, nullptr, nullptr, E);
  gemm_kernel<false, true><<<g1k, 256, 0, stream>>>(h1, E, Wvt, E, vcat, E, nullptr, nullptr, E);

  // 3) V -> V^T fragments source
  vtrans_kernel<<<dim3(S / 64, HB), 256, 0, stream>>>(vcat, vtb);

  // 4) causal attention -> concatenated heads (bf16 [B,S,E])
  attn_kernel<<<dim3(S / 128, HB), 256, 0, stream>>>(qcat, kcat, vtb, ocat);

  // 5) output projection + bias + residual(x) -> x1 (f32)
  gemm_kernel<false, false><<<g1k, 256, 0, stream>>>(ocat, E, pWt, E, x1, E, projb, x, E);

  // 6) LN2: x1 -> h2 (bf16)
  ln_kernel<<<BS, 256, 0, stream>>>(x1, ln2_g, ln2_b, h2, E);

  // 7) FFN: fin(relu), hid0(relu), hid1(relu), fout(+bias+residual x1)
  const dim3 gfd(D / 128, BS / 128, 1);
  gemm_kernel<true, true><<<gfd, 256, 0, stream>>>(h2, E, fiWt, E, f0, D, finb, nullptr, E);
  gemm_kernel<true, true><<<gfd, 256, 0, stream>>>(f0, D, hWt, D, f1, D, hidb, nullptr, D);
  gemm_kernel<true, true><<<gfd, 256, 0, stream>>>(f1, D, hWt + (size_t)D * D, D, f0, D, hidb + D, nullptr, D);
  gemm_kernel<false, false><<<g1k, 256, 0, stream>>>(f0, D, foWt, D, out, E, foutb, x1, D);
}
